// CMDPhenotypePredictor_28020366639639
// MI455X (gfx1250) — matrix operand order repaired and checked by reading, not compiled, not run on hardware
//
#include <hip/hip_runtime.h>
#include <hip/hip_bf16.h>

// ---------------- constants ----------------
#define B_SZ   16
#define L_OUT  2499          // conv front-end output length
#define MROWS  (B_SZ * L_OUT)        // 39984
#define MPAD   40064                 // 313 * 128
#define DMODEL 256
#define DIN    512
#define NZX    1288
#define NZXP   1408                  // padded to 128
#define NXBC   768
#define NHEADS 8
#define HDIM   64
#define DSTATE 128
#define NCHK   20                    // 2560 / 128
#define LPAD   2560
#define K1     19992
#define K1P    20000                 // padded to 32

typedef __attribute__((ext_vector_type(16))) _Float16 v16h;
typedef __attribute__((ext_vector_type(8)))  float    v8f;
typedef __attribute__((ext_vector_type(4)))  unsigned int v4u;
typedef __attribute__((ext_vector_type(8)))  int      v8i;
typedef __attribute__((ext_vector_type(4)))  int      v4i;

#if __has_builtin(__builtin_amdgcn_tensor_load_to_lds) && \
    __has_builtin(__builtin_amdgcn_s_wait_tensorcnt)
#define USE_TDM 1
#else
#define USE_TDM 0
#endif

// ---------------- WMMA helpers ----------------
__device__ __forceinline__ v8f wmma16(v16h a, v16h b, v8f c) {
  return __builtin_amdgcn_wmma_f32_16x16x32_f16(false, a, false, b, (short)0, c,
                                                false, false);
}

// A fragment: 16x32 f16, A row-major [M,K].  lane m = lane&15;
// halves 0..7 -> k = k0 + 8*(lane>>4) + h ; halves 8..15 -> +16.
__device__ __forceinline__ v16h frag_a(const _Float16* A, int ld, int m0, int k0) {
  unsigned lane = threadIdx.x & 31u;
  const _Float16* p = A + (size_t)(m0 + (int)(lane & 15u)) * ld + k0 + (int)((lane >> 4) << 3);
  v16h a;
  float4* ap = reinterpret_cast<float4*>(&a);
  ap[0] = *reinterpret_cast<const float4*>(p);
  ap[1] = *reinterpret_cast<const float4*>(p + 16);
  return a;
}

// B fragment (B = W^T, W row-major [N,K]): lane n = lane&15; halves 0..7 -> k = k0 + 8*(lane>>4) + h ; halves 8..15 -> +16
__device__ __forceinline__ v16h frag_bt(const _Float16* W, int ld, int n0, int k0) {
  unsigned lane = threadIdx.x & 31u;
  const _Float16* p = W + (size_t)(n0 + (int)(lane & 15u)) * ld + k0 + (int)((lane >> 4) << 3);
  v16h b;
  float4* bp = reinterpret_cast<float4*>(&b);
  bp[0] = *reinterpret_cast<const float4*>(p);
  bp[1] = *reinterpret_cast<const float4*>(p + 16);
  return b;
}

#if USE_TDM
// Issue a TDM load of a [128 rows x 32 cols] f16 tile (row stride ldK elements)
// from global into LDS at byte offset lds_off.  2D tensor descriptor (D#):
//   group0: count=1 | lds_addr | global_addr | type=2
//   group1: data_size=2B, tensor_dim0=32, tensor_dim1=128, tile=[32,128],
//           tensor_dim0_stride=ldK
__device__ __forceinline__ void tdm_load_128x32(const _Float16* g, int ldK,
                                                unsigned lds_off) {
  unsigned long long ga = (unsigned long long)(size_t)g;
  v4u g0;
  g0[0] = 1u;                                                   // count = 1
  g0[1] = lds_off;                                              // D#.lds_addr
  g0[2] = (unsigned)ga;                                         // global_addr lo
  g0[3] = (unsigned)((ga >> 32) & 0x01FFFFFFull) | (2u << 30);  // addr hi | type=2
  v8i g1;
  g1[0] = 1 << 16;          // workgroup_mask=0, data_size=1 (2 bytes)
  g1[1] = (int)(32u << 16); // tensor_dim0 = 32  (bits 63:48)
  g1[2] = (int)(128u << 16);// tensor_dim0 hi=0 | tensor_dim1 = 128
  g1[3] = (int)(32u << 16); // tensor_dim1 hi=0 | tile_dim0 = 32
  g1[4] = 128;              // tile_dim1 = 128, tile_dim2 = 0
  g1[5] = ldK;              // tensor_dim0_stride (low 32, in elements)
  g1[6] = 0;
  g1[7] = 0;
  v4i z;
  z[0] = 0; z[1] = 0; z[2] = 0; z[3] = 0;
  v8i z8;
  z8[0] = 0; z8[1] = 0; z8[2] = 0; z8[3] = 0;
  z8[4] = 0; z8[5] = 0; z8[6] = 0; z8[7] = 0;
  // 6-arg form (clang-23 / therock signature): g0, g1, g2, g3, extra group, cpol
  __builtin_amdgcn_tensor_load_to_lds(g0, g1, z, z, z8, 0);
}
#endif

// ---------------- generic WMMA GEMM:  C[M,N] = A[M,K] @ W[N,K]^T  -> f16 ----------------
// TDM path: wave 0 DMA-stages double-buffered 128x32 tiles of A and W into LDS,
// split-waits on TENSORcnt; all waves consume via ds_load_b128 fragments.
template <bool RELU>
__global__ void __launch_bounds__(256) gemm16_kernel(
    const _Float16* __restrict__ A, const _Float16* __restrict__ W,
    const float* __restrict__ bias, _Float16* __restrict__ C, int N, int K) {
  const unsigned lane = threadIdx.x & 31u;
  const int wv = threadIdx.x >> 5;                 // 8 waves: 4 (M) x 2 (N)
  const int mB = blockIdx.y * 128;
  const int nB = blockIdx.x * 128;
  const int mL = (wv & 3) * 32;
  const int nL = (wv >> 2) * 64;
  v8f acc[2][4];
#pragma unroll
  for (int i = 0; i < 2; ++i)
#pragma unroll
    for (int j = 0; j < 4; ++j)
#pragma unroll
      for (int r = 0; r < 8; ++r) acc[i][j][r] = 0.f;

#if USE_TDM
  extern __shared__ char gsm[];
  _Float16* sA = (_Float16*)gsm;          // [2][128*32]
  _Float16* sB = sA + 2 * 128 * 32;       // [2][128*32]
  const unsigned ldsA = (unsigned)(size_t)sA;
  const unsigned ldsB = (unsigned)(size_t)sB;
  const int S = K / 32;
  if (threadIdx.x < 32) {                 // wave 0 issues DMA for s = 0
    tdm_load_128x32(A + (size_t)mB * K, K, ldsA);
    tdm_load_128x32(W + (size_t)nB * K, K, ldsB);
  }
  for (int s = 0; s < S; ++s) {
    const int cur = s & 1;
    if (threadIdx.x < 32) {
      if (s + 1 < S) {
        const int nxt = (s + 1) & 1;
        tdm_load_128x32(A + (size_t)mB * K + (s + 1) * 32, K, ldsA + nxt * 8192u);
        tdm_load_128x32(W + (size_t)nB * K + (s + 1) * 32, K, ldsB + nxt * 8192u);
        __builtin_amdgcn_s_wait_tensorcnt((short)2);  // pair for s is done
      } else {
        __builtin_amdgcn_s_wait_tensorcnt((short)0);
      }
    }
    __syncthreads();                      // tile s visible to all waves
    const _Float16* tA = sA + cur * 4096;
    const _Float16* tB = sB + cur * 4096;
    v16h a0 = frag_a(tA, 32, mL, 0);
    v16h a1 = frag_a(tA, 32, mL + 16, 0);
#pragma unroll
    for (int j = 0; j < 4; ++j) {
      v16h vb = frag_bt(tB, 32, nL + j * 16, 0);
      acc[0][j] = wmma16(a0, vb, acc[0][j]);
      acc[1][j] = wmma16(a1, vb, acc[1][j]);
    }
    __syncthreads();                      // done reading before buffer reuse
  }
#else
  for (int k0 = 0; k0 < K; k0 += 32) {
    v16h a0 = frag_a(A, K, mB + mL, k0);
    v16h a1 = frag_a(A, K, mB + mL + 16, k0);
#pragma unroll
    for (int j = 0; j < 4; ++j) {
      v16h vb = frag_bt(W, K, nB + nL + j * 16, k0);
      acc[0][j] = wmma16(a0, vb, acc[0][j]);
      acc[1][j] = wmma16(a1, vb, acc[1][j]);
    }
  }
#endif

  const int mb = (int)((lane >> 4) << 3);
#pragma unroll
  for (int i = 0; i < 2; ++i)
#pragma unroll
    for (int j = 0; j < 4; ++j) {
      int nn = nB + nL + j * 16 + (int)(lane & 15u);
      float bv = bias ? bias[nn] : 0.f;
#pragma unroll
      for (int r = 0; r < 8; ++r) {
        int mm = mB + mL + i * 16 + mb + r;
        float v = acc[i][j][r] + bv;
        if (RELU) v = v > 0.f ? v : 0.f;
        C[(size_t)mm * N + nn] = (_Float16)v;
      }
    }
}

// ---------------- front end: conv -> BN stats -> BN+ReLU ----------------
__global__ void __launch_bounds__(256) conv0_kernel(const float* __restrict__ x,
                                                    const float* __restrict__ w,
                                                    const float* __restrict__ bias,
                                                    float* __restrict__ hout) {
  int idx = blockIdx.x * 256 + threadIdx.x;   // [row, c], row = b*2499+l
  int c = idx & 255;
  int r = idx >> 8;
  int b = r / L_OUT, l = r % L_OUT;
  const float* xb = x + (size_t)b * 4 * 20000 + l * 8;
  const float* wc = w + c * 64;
  float acc = bias[c];
#pragma unroll
  for (int ci = 0; ci < 4; ++ci)
#pragma unroll
    for (int k = 0; k < 16; ++k) acc += xb[ci * 20000 + k] * wc[ci * 16 + k];
  hout[idx] = acc;
}

__global__ void __launch_bounds__(256) bn_stats_kernel(const float* __restrict__ h,
                                                       const float* __restrict__ bn_w,
                                                       const float* __restrict__ bn_b,
                                                       float* __restrict__ scale,
                                                       float* __restrict__ shift) {
  int c = blockIdx.x;
  __shared__ float r1[256], r2[256];
  float s = 0.f, s2 = 0.f;
  for (int i = threadIdx.x; i < MROWS; i += 256) {
    float v = h[(size_t)i * 256 + c];
    s += v;
    s2 += v * v;
  }
  r1[threadIdx.x] = s;
  r2[threadIdx.x] = s2;
  __syncthreads();
  for (int off = 128; off > 0; off >>= 1) {
    if (threadIdx.x < (unsigned)off) {
      r1[threadIdx.x] += r1[threadIdx.x + off];
      r2[threadIdx.x] += r2[threadIdx.x + off];
    }
    __syncthreads();
  }
  if (threadIdx.x == 0) {
    float mean = r1[0] / (float)MROWS;
    float var = r2[0] / (float)MROWS - mean * mean;
    float sc = bn_w[c] * rsqrtf(var + 1e-5f);
    scale[c] = sc;
    shift[c] = bn_b[c] - mean * sc;
  }
}

__global__ void __launch_bounds__(256) bn_apply_kernel(const float* __restrict__ h,
                                                       const float* __restrict__ scale,
                                                       const float* __restrict__ shift,
                                                       _Float16* __restrict__ u16) {
  int idx = blockIdx.x * 256 + threadIdx.x;
  int c = idx & 255;
  float v = h[idx] * scale[c] + shift[c];
  u16[idx] = (_Float16)(v > 0.f ? v : 0.f);
}

// ---------------- weight convert (with zero padding) ----------------
__global__ void __launch_bounds__(256) convert_w_kernel(const float* __restrict__ src,
                                                        _Float16* __restrict__ dst,
                                                        int N, int K, int Np, int Kp) {
  int idx = blockIdx.x * 256 + threadIdx.x;
  if (idx >= Np * Kp) return;
  int n = idx / Kp, k = idx % Kp;
  float v = (n < N && k < K) ? src[(size_t)n * K + k] : 0.f;
  dst[idx] = (_Float16)v;
}

// ---------------- depthwise causal conv + SiLU ----------------
__global__ void __launch_bounds__(256) dwconv_kernel(const _Float16* __restrict__ zx,
                                                     const float* __restrict__ w,
                                                     const float* __restrict__ bias,
                                                     _Float16* __restrict__ xbc) {
  int idx = blockIdx.x * 256 + threadIdx.x;   // [row, ch], ch<768
  int ch = idx % NXBC;
  int r = idx / NXBC;
  int l = r % L_OUT;
  float acc = bias[ch];
#pragma unroll
  for (int j = 0; j < 4; ++j) {
    int ll = l - 3 + j;
    if (ll >= 0) acc += w[ch * 4 + j] * (float)zx[(size_t)(r - 3 + j) * NZXP + 512 + ch];
  }
  acc = acc / (1.f + __expf(-acc));
  xbc[idx] = (_Float16)acc;
}

// ---------------- SSD phase A: G = C.B^T, M = mask(G), Ydiag = M.Xw, S = Bdec^T.Xw ----------------
__global__ void __launch_bounds__(256) ssd_a_kernel(
    const _Float16* __restrict__ zx, const _Float16* __restrict__ xbc,
    const float* __restrict__ dt_bias, const float* __restrict__ A_log,
    float* __restrict__ AcsBuf, float* __restrict__ Ybuf, _Float16* __restrict__ Sbuf) {
  extern __shared__ char smem[];
  _Float16* sC = (_Float16*)smem;              // 128x128 (later aliased as sM)
  _Float16* sB = sC + 128 * 128;               // 128x128
  _Float16* sBdT = sB + 128 * 128;             // 128x128 (rows = n, cols = l)
  _Float16* sXwT = sBdT + 128 * 128;           // 64x128  (rows = p, cols = l)
  float* sdt = (float*)(sXwT + 64 * 128);      // 128
  float* sAcs = sdt + 128;                     // 128

  const int bid = blockIdx.x;
  const int h = bid & 7;
  const int c = (bid >> 3) % NCHK;
  const int b = bid / (NCHK * NHEADS);
  const int tid = threadIdx.x;
  const float Ah = -__expf(A_log[h]);

  if (tid < 128) {
    int gl = c * 128 + tid;
    float dtv = 0.f;
    if (gl < L_OUT) {
      float raw = (float)zx[(size_t)(b * L_OUT + gl) * NZXP + 1280 + h] + dt_bias[h];
      dtv = raw > 20.f ? raw : log1pf(__expf(raw));
    }
    sdt[tid] = dtv;
  }
  __syncthreads();
  if (tid == 0) {
    float run = 0.f;
    for (int i = 0; i < 128; ++i) {
      run += sdt[i] * Ah;
      sAcs[i] = run;
    }
  }
  __syncthreads();
  if (tid < 128) AcsBuf[(((size_t)(b * NHEADS + h) * NCHK + c) << 7) + tid] = sAcs[tid];
  const float alast = sAcs[127];

  for (int idx = tid; idx < 128 * 128; idx += 256) {
    int l = idx >> 7, n = idx & 127;
    int gl = c * 128 + l;
    float bv = 0.f, cv = 0.f;
    if (gl < L_OUT) {
      size_t r = (size_t)(b * L_OUT + gl) * NXBC;
      bv = (float)xbc[r + 512 + n];
      cv = (float)xbc[r + 640 + n];
    }
    sB[idx] = (_Float16)bv;
    sC[idx] = (_Float16)cv;
    sBdT[n * 128 + l] = (_Float16)(bv * __expf(alast - sAcs[l]));
  }
  for (int idx = tid; idx < 64 * 128; idx += 256) {
    int l = idx & 127, p = idx >> 7;
    int gl = c * 128 + l;
    float xv = 0.f;
    if (gl < L_OUT) xv = (float)xbc[(size_t)(b * L_OUT + gl) * NXBC + h * 64 + p] * sdt[l];
    sXwT[p * 128 + l] = (_Float16)xv;
  }
  __syncthreads();

  const int wv = tid >> 5;
  const unsigned lane = tid & 31u;
  const int m0 = wv * 16;
  const int mrow = m0 + (int)((lane >> 4) << 3);

  // G = C @ B^T  (16-row strip per wave, full 128 cols)
  v8f g[8];
#pragma unroll
  for (int j = 0; j < 8; ++j)
#pragma unroll
    for (int r = 0; r < 8; ++r) g[j][r] = 0.f;
  for (int k0 = 0; k0 < 128; k0 += 32) {
    v16h va = frag_a(sC, 128, m0, k0);
#pragma unroll
    for (int j = 0; j < 8; ++j) {
      v16h vb = frag_bt(sB, 128, j * 16, k0);
      g[j] = wmma16(va, vb, g[j]);
    }
  }
  __syncthreads();  // all G reads of sC done; safe to alias
  _Float16* sM = sC;
#pragma unroll
  for (int j = 0; j < 8; ++j)
#pragma unroll
    for (int r = 0; r < 8; ++r) {
      int mm = mrow + r;
      int nn = j * 16 + (int)(lane & 15u);
      float v = (nn <= mm) ? g[j][r] * __expf(sAcs[mm] - sAcs[nn]) : 0.f;
      sM[mm * 128 + nn] = (_Float16)v;
    }
  __syncthreads();

  // Ydiag = M @ Xw   and   S = BdT @ Xw
  v8f y[4], s[4];
#pragma unroll
  for (int j = 0; j < 4; ++j)
#pragma unroll
    for (int r = 0; r < 8; ++r) {
      y[j][r] = 0.f;
      s[j][r] = 0.f;
    }
  for (int k0 = 0; k0 < 128; k0 += 32) {
    v16h va = frag_a(sM, 128, m0, k0);
    v16h vs = frag_a(sBdT, 128, m0, k0);
#pragma unroll
    for (int j = 0; j < 4; ++j) {
      v16h vb = frag_bt(sXwT, 128, j * 16, k0);
      y[j] = wmma16(va, vb, y[j]);
      s[j] = wmma16(vs, vb, s[j]);
    }
  }
#pragma unroll
  for (int j = 0; j < 4; ++j) {
    int pp = j * 16 + (int)(lane & 15u);
#pragma unroll
    for (int r = 0; r < 8; ++r) {
      int q = mrow + r;
      Ybuf[(((size_t)b * LPAD + c * 128 + q) * NHEADS + h) * 64 + pp] = y[j][r];
      Sbuf[((((size_t)(b * NHEADS + h) * NCHK + c) * 128) + q) * 64 + pp] = (_Float16)s[j][r];
    }
  }
}

// ---------------- sequential inter-chunk scan ----------------
__global__ void __launch_bounds__(256) scan_kernel(const _Float16* __restrict__ S,
                                                   const float* __restrict__ Acs,
                                                   _Float16* __restrict__ prev) {
  int bh = blockIdx.x;  // b*8+h
  size_t base = (size_t)bh * NCHK * 8192;
  float carry[32];
#pragma unroll
  for (int j = 0; j < 32; ++j) carry[j] = 0.f;
  for (int c = 0; c < NCHK; ++c) {
    float cd = __expf(Acs[((size_t)bh * NCHK + c) * 128 + 127]);
    size_t cb = base + (size_t)c * 8192;
#pragma unroll
    for (int j = 0; j < 32; ++j) {
      int e = threadIdx.x + j * 256;
      prev[cb + e] = (_Float16)carry[j];
      carry[j] = carry[j] * cd + (float)S[cb + e];
    }
  }
}

// ---------------- SSD phase B: Yoff = (C * exp(Acs)) @ prev^T ; add D-skip ----------------
__global__ void __launch_bounds__(256) ssd_b_kernel(
    const _Float16* __restrict__ xbc, const _Float16* __restrict__ prev,
    const float* __restrict__ AcsBuf, const float* __restrict__ D,
    float* __restrict__ Ybuf) {
  extern __shared__ char smem[];
  _Float16* sCs = (_Float16*)smem;        // 128x128: C scaled by exp(Acs[l])
  _Float16* sPT = sCs + 128 * 128;        // 64x128:  rows = p, cols = n
  float* sA = (float*)(sPT + 64 * 128);   // 128

  const int bid = blockIdx.x;
  const int h = bid & 7;
  const int c = (bid >> 3) % NCHK;
  const int b = bid / (NCHK * NHEADS);
  const int tid = threadIdx.x;
  const size_t pbase = ((size_t)(b * NHEADS + h) * NCHK + c) * 8192;

  if (tid < 128) sA[tid] = AcsBuf[(((size_t)(b * NHEADS + h) * NCHK + c) << 7) + tid];
  __syncthreads();
  for (int idx = tid; idx < 128 * 128; idx += 256) {
    int l = idx >> 7, n = idx & 127;
    int gl = c * 128 + l;
    float cv = 0.f;
    if (gl < L_OUT)
      cv = (float)xbc[(size_t)(b * L_OUT + gl) * NXBC + 640 + n] * __expf(sA[l]);
    sCs[idx] = (_Float16)cv;
  }
  for (int idx = tid; idx < 64 * 128; idx += 256) {
    int p = idx >> 7, n = idx & 127;
    sPT[p * 128 + n] = prev[pbase + (size_t)n * 64 + p];
  }
  __syncthreads();

  const int wv = tid >> 5;
  const unsigned lane = tid & 31u;
  const int m0 = wv * 16;
  const int mrow = m0 + (int)((lane >> 4) << 3);
  v8f y[4];
#pragma unroll
  for (int j = 0; j < 4; ++j)
#pragma unroll
    for (int r = 0; r < 8; ++r) y[j][r] = 0.f;
  for (int k0 = 0; k0 < 128; k0 += 32) {
    v16h va = frag_a(sCs, 128, m0, k0);
#pragma unroll
    for (int j = 0; j < 4; ++j) {
      v16h vb = frag_bt(sPT, 128, j * 16, k0);
      y[j] = wmma16(va, vb, y[j]);
    }
  }
  const float Dv = D[h];
#pragma unroll
  for (int j = 0; j < 4; ++j) {
    int pp = j * 16 + (int)(lane & 15u);
#pragma unroll
    for (int r = 0; r < 8; ++r) {
      int q = mrow + r;
      int gl = c * 128 + q;
      if (gl < L_OUT) {
        size_t yi = (((size_t)b * LPAD + gl) * NHEADS + h) * 64 + pp;
        float xs = (float)xbc[(size_t)(b * L_OUT + gl) * NXBC + h * 64 + pp];
        Ybuf[yi] = Ybuf[yi] + y[j][r] + Dv * xs;
      }
    }
  }
}

// ---------------- gated RMSNorm ----------------
__global__ void __launch_bounds__(256) gate_rms_kernel(const float* __restrict__ Ybuf,
                                                       const _Float16* __restrict__ zx,
                                                       const float* __restrict__ norm_w,
                                                       _Float16* __restrict__ yout) {
  int r = blockIdx.x;  // 0..MROWS-1
  int b = r / L_OUT, l = r % L_OUT;
  __shared__ float red[256];
  float gv[2];
  float ss = 0.f;
#pragma unroll
  for (int t = 0; t < 2; ++t) {
    int d = threadIdx.x + t * 256;
    float yv = Ybuf[(((size_t)b * LPAD + l) * NHEADS + (d >> 6)) * 64 + (d & 63)];
    float zv = (float)zx[(size_t)r * NZXP + d];
    float g = yv * (zv / (1.f + __expf(-zv)));
    gv[t] = g;
    ss += g * g;
  }
  red[threadIdx.x] = ss;
  __syncthreads();
  for (int off = 128; off > 0; off >>= 1) {
    if (threadIdx.x < (unsigned)off) red[threadIdx.x] += red[threadIdx.x + off];
    __syncthreads();
  }
  float sc = rsqrtf(red[0] / (float)DIN + 1e-5f);
#pragma unroll
  for (int t = 0; t < 2; ++t) {
    int d = threadIdx.x + t * 256;
    yout[(size_t)r * DIN + d] = (_Float16)(gv[t] * sc * norm_w[d]);
  }
}

// ---------------- conv1 ("same" pad) -> flattened f16 features ----------------
__global__ void __launch_bounds__(256) conv1_kernel(const _Float16* __restrict__ yproj,
                                                    const float* __restrict__ w,
                                                    const float* __restrict__ bias,
                                                    _Float16* __restrict__ f) {
  int idx = blockIdx.x * 256 + threadIdx.x;
  if (idx >= B_SZ * 8 * L_OUT) return;
  int l = idx % L_OUT;
  int t = idx / L_OUT;
  int s = t & 7, b = t >> 3;
  float acc = bias[s];
  for (int k = 0; k < 16; ++k) {
    int ll = l + k - 7;
    if (ll < 0 || ll >= L_OUT) continue;
    const _Float16* yr = yproj + (size_t)(b * L_OUT + ll) * 256;
    __builtin_prefetch(yr + 256, 0, 0);     // global_prefetch_b8: next row
    const float* wr = w + (size_t)s * 256 * 16 + k;
    float a2 = 0.f;
    for (int cc = 0; cc < 256; ++cc) a2 += (float)yr[cc] * wr[cc * 16];
    acc += a2;
  }
  f[(size_t)b * K1P + s * L_OUT + l] = (_Float16)acc;
}

__global__ void fpad_kernel(_Float16* f) {
  int i = threadIdx.x;  // 128 = 16 rows x 8 pad cols
  if (i < 128) f[(size_t)(i >> 3) * K1P + K1 + (i & 7)] = (_Float16)0.f;
}

__global__ void b3pad_kernel(const float* b3, float* dst) {
  int i = threadIdx.x;
  if (i < 128) dst[i] = (i < 64) ? b3[i] : 0.f;
}

__global__ void head_kernel(const _Float16* __restrict__ a3, const float* __restrict__ w4,
                            const float* __restrict__ b4, float* __restrict__ out) {
  int b = threadIdx.x;
  if (b < B_SZ) {
    float acc = b4[0];
    for (int j = 0; j < 64; ++j) acc += (float)a3[b * 128 + j] * w4[j];
    out[b] = acc;
  }
}

// ---------------- host launcher ----------------
extern "C" void kernel_launch(void* const* d_in, const int* in_sizes, int n_in,
                              void* d_out, int out_size, void* d_ws, size_t ws_size,
                              hipStream_t stream) {
  const float* x        = (const float*)d_in[0];
  const float* conv_w   = (const float*)d_in[1];
  const float* conv_b   = (const float*)d_in[2];
  const float* bn_w     = (const float*)d_in[3];
  const float* bn_b     = (const float*)d_in[4];
  const float* in_proj  = (const float*)d_in[5];
  const float* c1d_w    = (const float*)d_in[6];
  const float* c1d_b    = (const float*)d_in[7];
  const float* dt_bias  = (const float*)d_in[8];
  const float* A_log    = (const float*)d_in[9];
  const float* Dp       = (const float*)d_in[10];
  const float* norm_w   = (const float*)d_in[11];
  const float* out_proj = (const float*)d_in[12];
  const float* conv1_w  = (const float*)d_in[13];
  const float* conv1_b  = (const float*)d_in[14];
  const float* w1 = (const float*)d_in[15]; const float* b1 = (const float*)d_in[16];
  const float* w2 = (const float*)d_in[17]; const float* b2 = (const float*)d_in[18];
  const float* w3 = (const float*)d_in[19]; const float* b3 = (const float*)d_in[20];
  const float* w4 = (const float*)d_in[21]; const float* b4 = (const float*)d_in[22];
  float* out = (float*)d_out;

  size_t off = 0;
  auto alloc = [&](size_t bytes) {
    off = (off + 255) & ~(size_t)255;
    size_t cur = off;
    off += bytes;
    return cur;
  };
  char* ws = (char*)d_ws;
  float*     h_conv  = (float*)    (ws + alloc((size_t)MROWS * 256 * 4));
  float*     scale   = (float*)    (ws + alloc(256 * 4));
  float*     shift   = (float*)    (ws + alloc(256 * 4));
  _Float16*  u16     = (_Float16*) (ws + alloc((size_t)MPAD * 256 * 2));
  _Float16*  wip16   = (_Float16*) (ws + alloc((size_t)NZXP * 256 * 2));
  _Float16*  zx16    = (_Float16*) (ws + alloc((size_t)MPAD * NZXP * 2));
  _Float16*  xbc16   = (_Float16*) (ws + alloc((size_t)MROWS * NXBC * 2));
  float*     AcsBuf  = (float*)    (ws + alloc((size_t)B_SZ * NHEADS * NCHK * 128 * 4));
  float*     Ybuf    = (float*)    (ws + alloc((size_t)B_SZ * LPAD * NHEADS * 64 * 4));
  _Float16*  S16     = (_Float16*) (ws + alloc((size_t)B_SZ * NHEADS * NCHK * 8192 * 2));
  _Float16*  prev16  = (_Float16*) (ws + alloc((size_t)B_SZ * NHEADS * NCHK * 8192 * 2));
  _Float16*  y16     = (_Float16*) (ws + alloc((size_t)MPAD * DIN * 2));
  _Float16*  wop16   = (_Float16*) (ws + alloc((size_t)256 * 512 * 2));
  _Float16*  yproj16 = (_Float16*) (ws + alloc((size_t)MPAD * 256 * 2));
  _Float16*  fbuf    = (_Float16*) (ws + alloc((size_t)128 * K1P * 2));
  _Float16*  w1_16   = (_Float16*) (ws + alloc((size_t)512 * K1P * 2));
  _Float16*  w2_16   = (_Float16*) (ws + alloc((size_t)256 * 512 * 2));
  _Float16*  w3_16   = (_Float16*) (ws + alloc((size_t)128 * 256 * 2));
  float*     b3pad   = (float*)    (ws + alloc(128 * 4));
  _Float16*  a1_16   = (_Float16*) (ws + alloc((size_t)128 * 512 * 2));
  _Float16*  a2_16   = (_Float16*) (ws + alloc((size_t)128 * 256 * 2));
  _Float16*  a3_16   = (_Float16*) (ws + alloc((size_t)128 * 128 * 2));
  (void)ws_size; (void)n_in; (void)in_sizes; (void)out_size;

  const size_t gemmSmem = 2u * 2u * 128u * 32u * 2u;  // 32 KB double-buffered tiles

  // front end
  conv0_kernel<<<MROWS, 256, 0, stream>>>(x, conv_w, conv_b, h_conv);
  bn_stats_kernel<<<256, 256, 0, stream>>>(h_conv, bn_w, bn_b, scale, shift);
  bn_apply_kernel<<<MROWS, 256, 0, stream>>>(h_conv, scale, shift, u16);

  // weight converts (padded f16)
  convert_w_kernel<<<(NZXP * 256 + 255) / 256, 256, 0, stream>>>(in_proj, wip16, NZX, 256, NZXP, 256);
  convert_w_kernel<<<(256 * 512 + 255) / 256, 256, 0, stream>>>(out_proj, wop16, 256, 512, 256, 512);
  convert_w_kernel<<<(512 * K1P + 255) / 256, 256, 0, stream>>>(w1, w1_16, 512, K1, 512, K1P);
  convert_w_kernel<<<(256 * 512 + 255) / 256, 256, 0, stream>>>(w2, w2_16, 256, 512, 256, 512);
  convert_w_kernel<<<(128 * 256 + 255) / 256, 256, 0, stream>>>(w3, w3_16, 64, 256, 128, 256);
  b3pad_kernel<<<1, 128, 0, stream>>>(b3, b3pad);

  // in_proj GEMM: [MPAD,256] x [NZXP,256]^T
  gemm16_kernel<false><<<dim3(NZXP / 128, MPAD / 128), 256, gemmSmem, stream>>>(
      u16, wip16, nullptr, zx16, NZXP, 256);

  // depthwise conv + SiLU
  dwconv_kernel<<<(MROWS * NXBC) / 256, 256, 0, stream>>>(zx16, c1d_w, c1d_b, xbc16);

  // SSD
  const size_t smemA = (size_t)(3 * 128 * 128 + 64 * 128) * 2 + 256 * 4;
  ssd_a_kernel<<<B_SZ * NCHK * NHEADS, 256, smemA, stream>>>(zx16, xbc16, dt_bias, A_log,
                                                             AcsBuf, Ybuf, S16);
  scan_kernel<<<B_SZ * NHEADS, 256, 0, stream>>>(S16, AcsBuf, prev16);
  const size_t smemB = (size_t)(128 * 128 + 64 * 128) * 2 + 128 * 4;
  ssd_b_kernel<<<B_SZ * NCHK * NHEADS, 256, smemB, stream>>>(xbc16, prev16, AcsBuf, Dp, Ybuf);

  // gate + RMSNorm
  gate_rms_kernel<<<MROWS, 256, 0, stream>>>(Ybuf, zx16, norm_w, y16);

  // out_proj GEMM: [MPAD,512] x [256,512]^T
  gemm16_kernel<false><<<dim3(256 / 128, MPAD / 128), 256, gemmSmem, stream>>>(
      y16, wop16, nullptr, yproj16, 256, 512);

  // conv1 + flatten
  conv1_kernel<<<(B_SZ * 8 * L_OUT + 255) / 256, 256, 0, stream>>>(yproj16, conv1_w, conv1_b, fbuf);
  fpad_kernel<<<1, 128, 0, stream>>>(fbuf);

  // MLP
  gemm16_kernel<true><<<dim3(4, 1), 256, gemmSmem, stream>>>(fbuf, w1_16, b1, a1_16, 512, K1P);
  gemm16_kernel<true><<<dim3(2, 1), 256, gemmSmem, stream>>>(a1_16, w2_16, b2, a2_16, 256, 512);
  gemm16_kernel<true><<<dim3(1, 1), 256, gemmSmem, stream>>>(a2_16, w3_16, b3pad, a3_16, 128, 256);
  head_kernel<<<1, 64, 0, stream>>>(a3_16, w4, b4, out);
}
